// AdaptedMixer_57148834840870
// MI455X (gfx1250) — compile-verified, matrix operand order checked by reading, not run on hardware
//
#include <hip/hip_runtime.h>
#include <math.h>

// ---------------- problem constants ----------------
#define B_SZ    2
#define S_LEN   2048            // power of two: s = m & 2047, ch = idx >> 11
#define D_MODEL 768
#define I_DIM   1536
#define N_STATE 16
#define R_DIM   48
#define RP      64              // R padded to WMMA K granularity
#define SSM_E   (R_DIM + 2*N_STATE)   // 80
#define M_ROWS  (B_SZ * S_LEN)        // 4096
#define LDSROW  80              // padded LDS row stride (elems) for 64-elem K chunk

typedef __attribute__((ext_vector_type(16))) __bf16 v16bf;
typedef __attribute__((ext_vector_type(8)))  float  v8f;
typedef int v4i_t __attribute__((ext_vector_type(4)));
typedef int v2i_t __attribute__((ext_vector_type(2)));

#define WMMA_BF16(a, b, c) \
  __builtin_amdgcn_wmma_f32_16x16x32_bf16(false, (a), false, (b), (short)0, (c), false, false)

// ---------------- fragment load (ISA 7.12.2 bf16 A/B layout) ----------------
// lane L -> row r = L&15, khalf = L>>4. VGPR0..3 = K khalf*8..+7, VGPR4..7 = K 16+khalf*8..+7.
__device__ __forceinline__ v16bf load_frag16(const __bf16* rowbase, int kh) {
  union { v16bf v; uint4 q[2]; } f;
  f.q[0] = *reinterpret_cast<const uint4*>(rowbase + kh * 8);
  f.q[1] = *reinterpret_cast<const uint4*>(rowbase + 16 + kh * 8);
  return f.v;
}

// ---------------- async global->LDS copies (ASYNCcnt path) ----------------
__device__ __forceinline__ void async_cp_b128(const __bf16* g, __bf16* l) {
#if __has_builtin(__builtin_amdgcn_global_load_async_to_lds_b128)
  __builtin_amdgcn_global_load_async_to_lds_b128(
      (__attribute__((address_space(1))) v4i_t*)g,
      (__attribute__((address_space(3))) v4i_t*)l, 0, 0);
#else
  unsigned loff = (unsigned)(__SIZE_TYPE__)(__attribute__((address_space(3))) void*)l;
  asm volatile("global_load_async_to_lds_b128 %0, %1, off"
               :: "v"(loff), "v"(g) : "memory");
#endif
}
__device__ __forceinline__ void async_cp_b64(const __bf16* g, __bf16* l) {
#if __has_builtin(__builtin_amdgcn_global_load_async_to_lds_b64)
  __builtin_amdgcn_global_load_async_to_lds_b64(
      (__attribute__((address_space(1))) v2i_t*)g,
      (__attribute__((address_space(3))) v2i_t*)l, 0, 0);
#else
  unsigned loff = (unsigned)(__SIZE_TYPE__)(__attribute__((address_space(3))) void*)l;
  asm volatile("global_load_async_to_lds_b64 %0, %1, off"
               :: "v"(loff), "v"(g) : "memory");
#endif
}

template<int N>
__device__ __forceinline__ void wait_async_le() {
#if __has_builtin(__builtin_amdgcn_s_wait_asynccnt)
  __builtin_amdgcn_s_wait_asynccnt((unsigned short)N);
#else
  asm volatile("s_wait_asynccnt %0" :: "i"(N) : "memory");
#endif
}

// Stage one W chunk (NB*16 rows x 64 k, bf16) into LDS (padded rows of LDSROW elems).
// Packed chunk = NB*16*128 bytes; uniform async ops per thread/wave.
template<int NB>
__device__ __forceinline__ void stage_async(const __bf16* __restrict__ Wc, int ldw,
                                            __bf16* buf) {
  const int tid = threadIdx.x;
#pragma unroll
  for (int p = 0; p < 2; ++p) {
    const int o   = p * 4096 + tid * 16;   // byte offset in packed chunk
    const int n   = o >> 7;                // 128 packed bytes per row
    const int off = (o & 127) >> 1;        // elem offset within row
    async_cp_b128(Wc + (size_t)n * ldw + off, buf + n * LDSROW + off);
  }
  if (NB == 5) {                           // 80-row chunk: 2048-byte tail via b64
    const int o   = 8192 + tid * 8;
    const int n   = o >> 7;
    const int off = (o & 127) >> 1;
    async_cp_b64(Wc + (size_t)n * ldw + off, buf + n * LDSROW + off);
  }
}

// ---------------- LDS-staged, double-buffered WMMA GEMM ----------------
// One wave: 16 x (16*NB) tile of C = A[M,K] * W[N,K]^T. Block: 8 waves share W via LDS.
template<int NB>
__device__ __forceinline__ void gemm_lds(const __bf16* __restrict__ A, int lda,
                                         const __bf16* __restrict__ W, int ldw,
                                         int n0, int K, __bf16* lds, v8f acc[NB]) {
  constexpr int CHUNK = NB * 16 * LDSROW;
  constexpr int OPS   = (NB == 5) ? 3 : 2;   // async ops per wave per chunk
  const int wave = threadIdx.x >> 5;
  const int lane = threadIdx.x & 31;
  const int r = lane & 15, kh = lane >> 4;
  const int m0 = blockIdx.x * 128 + wave * 16;
  const __bf16* arow  = A + (size_t)(m0 + r) * lda;
  const __bf16* wbase = W + (size_t)n0 * ldw;
  const int nch = K >> 6;

  stage_async<NB>(wbase, ldw, lds);            // chunk 0 -> buf 0
  v16bf a0 = load_frag16(arow, kh);
  v16bf a1 = load_frag16(arow + 32, kh);

  for (int c = 0; c < nch; ++c) {
    __bf16* cur = lds + (c & 1) * CHUNK;
    v16bf na0 = a0, na1 = a1;
    if (c + 1 < nch) {                         // prefetch next chunk (W async, A regs)
      stage_async<NB>(wbase + (c + 1) * 64, ldw, lds + ((c + 1) & 1) * CHUNK);
      na0 = load_frag16(arow + (c + 1) * 64, kh);
      na1 = load_frag16(arow + (c + 1) * 64 + 32, kh);
      wait_async_le<OPS>();                    // retire chunk c, keep c+1 in flight
    } else {
      wait_async_le<0>();
    }
    __syncthreads();

    v16bf b0[NB], b1[NB];
#pragma unroll
    for (int j = 0; j < NB; ++j) {
      const __bf16* rb = cur + (j * 16 + r) * LDSROW;
      b0[j] = load_frag16(rb, kh);
      b1[j] = load_frag16(rb + 32, kh);
    }
#pragma unroll
    for (int j = 0; j < NB; ++j) acc[j] = WMMA_BF16(a0, b0[j], acc[j]);
#pragma unroll
    for (int j = 0; j < NB; ++j) acc[j] = WMMA_BF16(a1, b1[j], acc[j]);

    __syncthreads();                           // protect cur before chunk c+2 overwrites
    a0 = na0; a1 = na1;
  }
}

__device__ __forceinline__ float siluf(float x) { return x / (1.0f + expf(-x)); }
__device__ __forceinline__ float softplusf(float x) {
  return (x > 20.0f) ? x : log1pf(expf(x));
}

// ---------------- conversion kernels ----------------
__global__ void k_cvt(const float* __restrict__ in, __bf16* __restrict__ out, int n) {
  int i = blockIdx.x * blockDim.x + threadIdx.x;
  int stride = gridDim.x * blockDim.x;
  for (; i < n; i += stride) out[i] = (__bf16)in[i];
}

// dt_proj_w [I,48] -> bf16 [I,64] with zero pad
__global__ void k_cvt_dtw(const float* __restrict__ w, __bf16* __restrict__ out) {
  int i = blockIdx.x * blockDim.x + threadIdx.x;
  if (i >= I_DIM * RP) return;
  int r = i >> 6, c = i & 63;
  out[i] = (c < R_DIM) ? (__bf16)w[r * R_DIM + c] : (__bf16)0.0f;
}

// ---------------- 1) in_proj GEMM, epilogue splits into hidden/gate [B,I,S] ----
__global__ __launch_bounds__(256)
void k_in_proj(const __bf16* __restrict__ Abf, const __bf16* __restrict__ Wbf,
               float* __restrict__ hidden, float* __restrict__ gate) {
  __shared__ __align__(16) __bf16 smem[2 * 4 * 16 * LDSROW];
  const int n0 = blockIdx.y * 64;
  v8f acc[4] = {};
  gemm_lds<4>(Abf, D_MODEL, Wbf, D_MODEL, n0, D_MODEL, smem, acc);

  const int lane = threadIdx.x & 31;
  const int nl = lane & 15, kh = lane >> 4;
  const int m0 = blockIdx.x * 128 + (threadIdx.x >> 5) * 16;
#pragma unroll
  for (int j = 0; j < 4; ++j) {
    const int n = n0 + j * 16 + nl;
#pragma unroll
    for (int v = 0; v < 8; ++v) {
      const int m = m0 + v + 8 * kh;
      const int b = m >> 11, s = m & (S_LEN - 1);
      const float val = acc[j][v];
      if (n < I_DIM)
        hidden[((size_t)(b * I_DIM + n)) * S_LEN + s] = val;
      else
        gate[((size_t)(b * I_DIM + (n - I_DIM))) * S_LEN + s] = val;
    }
  }
}

// ---------------- 2) causal depthwise conv (K=4) + SiLU ----------------
__global__ void k_conv(const float* __restrict__ hidden, const float* __restrict__ cw,
                       const float* __restrict__ cb, float* __restrict__ h_bis,
                       __bf16* __restrict__ h_bsi) {
  size_t idx = (size_t)blockIdx.x * blockDim.x + threadIdx.x;
  if (idx >= (size_t)B_SZ * I_DIM * S_LEN) return;
  const int s  = (int)(idx & (S_LEN - 1));
  const int ch = (int)(idx >> 11);          // b*I + i
  const int i  = ch % I_DIM;
  const int b  = ch / I_DIM;
  const float* hp = hidden + (size_t)ch * S_LEN;
  float acc = cb[i];
#pragma unroll
  for (int k = 0; k < 4; ++k) {
    const int sp = s - 3 + k;
    if (sp >= 0) acc += hp[sp] * cw[i * 4 + k];
  }
  const float hv = siluf(acc);
  h_bis[idx] = hv;
  h_bsi[((size_t)(b * S_LEN + s)) * I_DIM + i] = (__bf16)hv;
}

// ---------------- 3) x_proj GEMM (N=80), emits ssm_p f32 + padded bf16 ts ----
__global__ __launch_bounds__(256)
void k_x_proj(const __bf16* __restrict__ hbf, const __bf16* __restrict__ Wbf,
              float* __restrict__ ssm, __bf16* __restrict__ tsbf) {
  __shared__ __align__(16) __bf16 smem[2 * 5 * 16 * LDSROW];
  v8f acc[5] = {};
  gemm_lds<5>(hbf, I_DIM, Wbf, I_DIM, 0, I_DIM, smem, acc);

  const int lane = threadIdx.x & 31;
  const int nl = lane & 15, kh = lane >> 4;
  const int m0 = blockIdx.x * 128 + (threadIdx.x >> 5) * 16;
#pragma unroll
  for (int j = 0; j < 5; ++j) {
    const int n = j * 16 + nl;
#pragma unroll
    for (int v = 0; v < 8; ++v) {
      const int m = m0 + v + 8 * kh;
      const float val = acc[j][v];
      ssm[(size_t)m * SSM_E + n] = val;
      if (n < R_DIM)      tsbf[(size_t)m * RP + n] = (__bf16)val;
      else if (n < RP)    tsbf[(size_t)m * RP + n] = (__bf16)0.0f;   // zero pad
    }
  }
}

// ---------------- 4) dt GEMM + bias + softplus (+alpha last step) -> [B,I,S] ----
__global__ __launch_bounds__(256)
void k_dt(const __bf16* __restrict__ tsbf, const __bf16* __restrict__ Wdt,
          const float* __restrict__ bias, const float* __restrict__ alpha,
          float* __restrict__ dt_bis) {
  __shared__ __align__(16) __bf16 smem[2 * 4 * 16 * LDSROW];
  const int n0 = blockIdx.y * 64;
  v8f acc[4] = {};
  gemm_lds<4>(tsbf, RP, Wdt, RP, n0, RP, smem, acc);

  const int lane = threadIdx.x & 31;
  const int nl = lane & 15, kh = lane >> 4;
  const int m0 = blockIdx.x * 128 + (threadIdx.x >> 5) * 16;
#pragma unroll
  for (int j = 0; j < 4; ++j) {
    const int n = n0 + j * 16 + nl;               // channel i
#pragma unroll
    for (int v = 0; v < 8; ++v) {
      const int m = m0 + v + 8 * kh;
      const int b = m >> 11, s = m & (S_LEN - 1);
      float sp = softplusf(acc[j][v] + bias[n]);
      if (s == S_LEN - 1) sp *= alpha[n];
      dt_bis[((size_t)(b * I_DIM + n)) * S_LEN + s] = sp;
    }
  }
}

// ---------------- 5) selective scan: lane = state index n, 2 ch per wave32 ----
__global__ __launch_bounds__(256)
void k_scan(const float* __restrict__ dt_bis, const float* __restrict__ h_bis,
            const float* __restrict__ ssm, const float* __restrict__ A_log,
            float* __restrict__ ybuf) {
  const int half = threadIdx.x >> 4;              // 0..15 in block of 256
  const int n    = threadIdx.x & 15;
  const int ch   = blockIdx.x * 16 + half;        // b*I + i
  const int b    = ch / I_DIM;
  const int i    = ch % I_DIM;

  const float a = -expf(A_log[i * N_STATE + n]);
  const float* dtp = dt_bis + (size_t)ch * S_LEN;
  const float* hp  = h_bis  + (size_t)ch * S_LEN;
  const float* sp  = ssm + (size_t)b * S_LEN * SSM_E;
  float* yp = ybuf + (size_t)ch * S_LEN;

  float st = 0.0f;
  for (int s = 0; s < S_LEN; ++s) {
    const float dtv = dtp[s];
    const float hv  = hp[s];
    const float* row = sp + (size_t)s * SSM_E;
    const float Bv = row[R_DIM + n];
    const float Cv = row[R_DIM + N_STATE + n];
    st = expf(a * dtv) * st + dtv * Bv * hv;
    float y = st * Cv;                             // 16-lane butterfly reduce
    y += __shfl_xor(y, 1, 32);
    y += __shfl_xor(y, 2, 32);
    y += __shfl_xor(y, 4, 32);
    y += __shfl_xor(y, 8, 32);
    if (n == 0) yp[s] = y;
  }
}

// ---------------- 6) gating elementwise, emit bf16 [B,S,I] ----------------
__global__ void k_gate(const float* __restrict__ ybuf, const float* __restrict__ h_bis,
                       const float* __restrict__ gate, const float* __restrict__ Dp,
                       const float* __restrict__ fg, __bf16* __restrict__ ybf) {
  size_t idx = (size_t)blockIdx.x * blockDim.x + threadIdx.x;
  if (idx >= (size_t)B_SZ * I_DIM * S_LEN) return;
  const int s  = (int)(idx & (S_LEN - 1));
  const int ch = (int)(idx >> 11);
  const int i  = ch % I_DIM;
  const int b  = ch / I_DIM;
  const float g = gate[idx];
  float val = (ybuf[idx] + h_bis[idx] * Dp[i]) * siluf(g);
  if (s == S_LEN - 1) val *= fg[i];
  ybf[((size_t)(b * S_LEN + s)) * I_DIM + i] = (__bf16)val;
}

// ---------------- 7) out_proj GEMM -> f32 output [B,S,Dm] ----------------
__global__ __launch_bounds__(256)
void k_out(const __bf16* __restrict__ ybf, const __bf16* __restrict__ Wbf,
           float* __restrict__ out) {
  __shared__ __align__(16) __bf16 smem[2 * 4 * 16 * LDSROW];
  const int n0 = blockIdx.y * 64;
  v8f acc[4] = {};
  gemm_lds<4>(ybf, I_DIM, Wbf, I_DIM, n0, I_DIM, smem, acc);

  const int lane = threadIdx.x & 31;
  const int nl = lane & 15, kh = lane >> 4;
  const int m0 = blockIdx.x * 128 + (threadIdx.x >> 5) * 16;
#pragma unroll
  for (int j = 0; j < 4; ++j) {
    const int n = n0 + j * 16 + nl;
#pragma unroll
    for (int v = 0; v < 8; ++v) {
      const int m = m0 + v + 8 * kh;
      out[(size_t)m * D_MODEL + n] = acc[j][v];
    }
  }
}

// ---------------- host launch ----------------
extern "C" void kernel_launch(void* const* d_in, const int* in_sizes, int n_in,
                              void* d_out, int out_size, void* d_ws, size_t ws_size,
                              hipStream_t stream) {
  const float* input      = (const float*)d_in[0];
  const float* in_proj_w  = (const float*)d_in[1];
  const float* conv_w     = (const float*)d_in[2];
  const float* conv_b     = (const float*)d_in[3];
  const float* x_proj_w   = (const float*)d_in[4];
  const float* dt_proj_w  = (const float*)d_in[5];
  const float* dt_proj_b  = (const float*)d_in[6];
  const float* A_log      = (const float*)d_in[7];
  const float* Dvec       = (const float*)d_in[8];
  const float* out_proj_w = (const float*)d_in[9];
  const float* alpha      = (const float*)d_in[10];
  const float* fgate      = (const float*)d_in[11];
  float* out = (float*)d_out;

  char* ws = (char*)d_ws;
  size_t off = 0;
  auto carve = [&](size_t bytes) -> void* {
    void* p = ws + off;
    off = (off + bytes + 255) & ~(size_t)255;
    return p;
  };
  __bf16* inp_bf   = (__bf16*)carve((size_t)M_ROWS * D_MODEL * 2);
  __bf16* w_in_bf  = (__bf16*)carve((size_t)2 * I_DIM * D_MODEL * 2);
  __bf16* w_x_bf   = (__bf16*)carve((size_t)SSM_E * I_DIM * 2);
  __bf16* w_dt_bf  = (__bf16*)carve((size_t)I_DIM * RP * 2);
  __bf16* w_out_bf = (__bf16*)carve((size_t)D_MODEL * I_DIM * 2);
  float*  hidden   = (float*)carve((size_t)B_SZ * I_DIM * S_LEN * 4);
  float*  gate     = (float*)carve((size_t)B_SZ * I_DIM * S_LEN * 4);
  float*  h_bis    = (float*)carve((size_t)B_SZ * I_DIM * S_LEN * 4);
  __bf16* h_bsi    = (__bf16*)carve((size_t)M_ROWS * I_DIM * 2);
  float*  ssm      = (float*)carve((size_t)M_ROWS * SSM_E * 4);
  __bf16* ts_bf    = (__bf16*)carve((size_t)M_ROWS * RP * 2);
  float*  dt_bis   = (float*)carve((size_t)B_SZ * I_DIM * S_LEN * 4);
  float*  ybuf     = (float*)carve((size_t)B_SZ * I_DIM * S_LEN * 4);
  (void)ws_size; (void)in_sizes; (void)n_in; (void)out_size;

  auto cvt = [&](const float* src, __bf16* dst, int n) {
    hipLaunchKernelGGL(k_cvt, dim3((n + 255) / 256), dim3(256), 0, stream, src, dst, n);
  };
  cvt(input,      inp_bf,   M_ROWS * D_MODEL);
  cvt(in_proj_w,  w_in_bf,  2 * I_DIM * D_MODEL);
  cvt(x_proj_w,   w_x_bf,   SSM_E * I_DIM);
  cvt(out_proj_w, w_out_bf, D_MODEL * I_DIM);
  hipLaunchKernelGGL(k_cvt_dtw, dim3((I_DIM * RP + 255) / 256), dim3(256), 0, stream,
                     dt_proj_w, w_dt_bf);

  const int elem_blocks = (int)(((size_t)B_SZ * I_DIM * S_LEN + 255) / 256);

  hipLaunchKernelGGL(k_in_proj, dim3(M_ROWS / 128, (2 * I_DIM) / 64), dim3(256), 0, stream,
                     inp_bf, w_in_bf, hidden, gate);
  hipLaunchKernelGGL(k_conv, dim3(elem_blocks), dim3(256), 0, stream,
                     hidden, conv_w, conv_b, h_bis, h_bsi);
  hipLaunchKernelGGL(k_x_proj, dim3(M_ROWS / 128, 1), dim3(256), 0, stream,
                     h_bsi, w_x_bf, ssm, ts_bf);
  hipLaunchKernelGGL(k_dt, dim3(M_ROWS / 128, I_DIM / 64), dim3(256), 0, stream,
                     ts_bf, w_dt_bf, dt_proj_b, alpha, dt_bis);
  hipLaunchKernelGGL(k_scan, dim3((B_SZ * I_DIM) / 16), dim3(256), 0, stream,
                     dt_bis, h_bis, ssm, A_log, ybuf);
  hipLaunchKernelGGL(k_gate, dim3(elem_blocks), dim3(256), 0, stream,
                     ybuf, h_bis, gate, Dvec, fgate, h_bsi /*reuse as bf16 y*/);
  hipLaunchKernelGGL(k_out, dim3(M_ROWS / 128, D_MODEL / 64), dim3(256), 0, stream,
                     h_bsi, w_out_bf, out);
}
